// SAttention_49898930045404
// MI455X (gfx1250) — compile-verified, matrix operand order checked by reading, not run on hardware
//
#include <hip/hip_runtime.h>

// ---------------------------------------------------------------------------
// Stock self-attention block for MI455X (gfx1250, wave32, WMMA + TDM).
// GEMMs: v_wmma_f32_16x16x32_f16, B-panels staged to LDS via the Tensor Data
// Mover (tensor_load_to_lds, TENSORcnt). Problem is L2-resident (192MB);
// compute-bound on the matrix pipes.
// ---------------------------------------------------------------------------

#define NTOK   4096
#define DMODEL 512
#define NHEAD  8
#define DHEAD  64
#define DEMB   64
#define DQK    (DMODEL + DEMB)   // 576

typedef __attribute__((ext_vector_type(16))) _Float16 v16h;
typedef __attribute__((ext_vector_type(8)))  _Float16 v8h;
typedef __attribute__((ext_vector_type(8)))  float    v8f;
typedef __attribute__((ext_vector_type(4)))  unsigned int v4u;
typedef __attribute__((ext_vector_type(4)))  int      v4i;
typedef __attribute__((ext_vector_type(8)))  int      v8i;

// A-fragment (and B-fragment, on the transposed matrix) loader.
// Per ISA 7.12.2 (16-bit A 16x32): lane L holds row L%16; K chunk base
// (L/16)*8 covers K = base..base+7 (VGPRs 0..3) and K = base+16..base+23
// (VGPRs 4..7) -> two contiguous 16-byte loads. Works for global and LDS.
__device__ __forceinline__ v16h ldfrag(const _Float16* p) {
  union { v16h v; v8h h[2]; } u;
  u.h[0] = *(const v8h*)(p);
  u.h[1] = *(const v8h*)(p + 16);
  return u.v;
}

__device__ __forceinline__ v8f wmma_f16(v16h a, v16h b, v8f c) {
  return __builtin_amdgcn_wmma_f32_16x16x32_f16(false, a, false, b,
                                                (short)0, c, false, false);
}

// ---------------------------------------------------------------------------
// TDM: DMA a 2D f16 panel (rows x elems_per_row, row stride in elements)
// from global memory into LDS. D# bitfields per CDNA5 ISA ch.8:
//   group0: [1:0]=count=1, [63:32]=lds_addr, [120:64]=global_addr, [127:126]=2
//   group1: [17:16]=data_size(2B), [79:48]=tensor_dim0, [111:80]=tensor_dim1,
//           [127:112]=tile_dim0, [143:128]=tile_dim1, [207:160]=dim0_stride
// Groups 2/3 unused for a 2D tile (zeroed).
// ---------------------------------------------------------------------------
__device__ __forceinline__ void tdm_load_2d(_Float16* lds_dst, const _Float16* gsrc,
                                            unsigned int elems_per_row,
                                            unsigned int rows,
                                            unsigned int row_stride_elems)
{
  const unsigned long long ga  = (unsigned long long)(uintptr_t)gsrc;
  const unsigned int       lds = (unsigned int)(uintptr_t)lds_dst;

  union { unsigned long long q[2]; v4u v; } G0;
  G0.q[0] = 1ull | ((unsigned long long)lds << 32);                 // count=1, lds_addr
  G0.q[1] = (ga & 0x01FFFFFFFFFFFFFFull) | (2ull << 62);            // global_addr, type=2

  union { unsigned long long q[4]; v8i v; } G1;
  G1.q[0] = (1ull << 16)                                            // data_size = 2 bytes
          | ((unsigned long long)(elems_per_row & 0xFFFFu) << 48);  // tensor_dim0 lo
  G1.q[1] = (unsigned long long)((elems_per_row >> 16) & 0xFFFFu)   // tensor_dim0 hi
          | ((unsigned long long)rows << 16)                        // tensor_dim1
          | ((unsigned long long)(elems_per_row & 0xFFFFu) << 48);  // tile_dim0
  G1.q[2] = (unsigned long long)(rows & 0xFFFFu)                    // tile_dim1
          | ((unsigned long long)row_stride_elems << 32);           // dim0_stride lo32
  G1.q[3] = 0ull;                                                   // dim0_stride hi=0

  v4i z4 = {0, 0, 0, 0};
  v8i z8 = {0, 0, 0, 0, 0, 0, 0, 0};
  __builtin_amdgcn_tensor_load_to_lds(G0.v, G1.v, z4, z4, z8, 0);
}

// ---------------------------------------------------------------------------
// LayerNorm: one block per row, blockDim == D (64 or 512).
// ---------------------------------------------------------------------------
__global__ void ln_kernel(const float* __restrict__ in, int D,
                          const float* __restrict__ g, const float* __restrict__ b,
                          _Float16* __restrict__ out_h, int ldh,
                          float* __restrict__ out_f)
{
  __shared__ float s1[512];
  __shared__ float s2[512];
  const int row = blockIdx.x;
  const int t = threadIdx.x;
  float v = in[(size_t)row * D + t];
  s1[t] = v; s2[t] = v * v;
  __syncthreads();
  for (int s = D >> 1; s > 0; s >>= 1) {
    if (t < s) { s1[t] += s1[t + s]; s2[t] += s2[t + s]; }
    __syncthreads();
  }
  const float mu  = s1[0] / (float)D;
  const float var = s2[0] / (float)D - mu * mu;
  const float r   = rsqrtf(var + 1e-5f);
  const float y   = (v - mu) * r * g[t] + b[t];
  out_h[(size_t)row * ldh + t] = (_Float16)y;
  if (out_f) out_f[(size_t)row * D + t] = y;
}

// ---------------------------------------------------------------------------
// Weight convert + transpose: W [in,out] f32 -> Wt [out,in] f16.
// ---------------------------------------------------------------------------
__global__ void wcvt_kernel(const float* __restrict__ W, _Float16* __restrict__ Wt,
                            int in_dim, int out_dim)
{
  int idx = blockIdx.x * blockDim.x + threadIdx.x;
  if (idx >= in_dim * out_dim) return;
  int o = idx % out_dim, i = idx / out_dim;
  Wt[(size_t)o * in_dim + i] = (_Float16)W[idx];
}

// ---------------------------------------------------------------------------
// WMMA GEMM: C[M,N] = A[M,K] @ Bt[N,K]^T (+bias, relu, residual).
// Block = 8 waves, block tile 128(M) x 64(N); each wave owns a 16x64 strip
// (4 accumulators share one A-fragment per k-step -> 4x A reuse).
// The 64xK B-panel (<=72KB) is TDM-DMA'd into LDS once per block and shared
// by all waves as ds_load_b128 fragments. Requires ldb == K (true for all
// callers: weights are stored transposed, densely).
// ---------------------------------------------------------------------------
__global__ __launch_bounds__(256)
void gemm_kernel(const _Float16* __restrict__ A, int lda,
                 const _Float16* __restrict__ Bt, int K,
                 const float* __restrict__ bias,
                 const float* __restrict__ residual,
                 float* __restrict__ Cf,
                 _Float16* __restrict__ Ch, int ldc,
                 int trans_store, int relu)
{
  extern __shared__ _Float16 Bs[];           // [64][K]
  const int lane = threadIdx.x & 31, wid = threadIdx.x >> 5;
  const int tM = blockIdx.x * 128 + wid * 16;
  const int tN = blockIdx.y * 64;
  const int row = lane & 15, half = lane >> 4;

  if (wid == 0) {
    tdm_load_2d(Bs, Bt + (size_t)tN * K, (unsigned)K, 64u, (unsigned)K);
    __builtin_amdgcn_s_wait_tensorcnt(0);
  }
  __syncthreads();

  const _Float16* Ap = A  + (size_t)(tM + row) * lda + half * 8;
  const _Float16* Bp = Bs + (size_t)row * K + half * 8;

  v8f acc[4] = {};
  for (int k0 = 0; k0 < K; k0 += 32) {
    const v16h a = ldfrag(Ap + k0);
    #pragma unroll
    for (int jn = 0; jn < 4; ++jn)
      acc[jn] = wmma_f16(a, ldfrag(Bp + (size_t)jn * 16 * K + k0), acc[jn]);
  }

  // C layout: VGPR j holds element (m = (lane/16)*8 + j, n = lane%16).
  const int m0 = tM + half * 8;
  #pragma unroll
  for (int jn = 0; jn < 4; ++jn) {
    const int n = tN + jn * 16 + row;
    const float bv = bias ? bias[n] : 0.0f;
    #pragma unroll
    for (int j = 0; j < 8; ++j) {
      float v = acc[jn][j] + bv;
      if (relu) v = fmaxf(v, 0.0f);
      const int m = m0 + j;
      if (Cf) {
        float r = residual ? residual[(size_t)m * ldc + n] : 0.0f;
        Cf[(size_t)m * ldc + n] = r + v;
      }
      if (Ch) {
        if (trans_store) Ch[(size_t)n * ldc + m] = (_Float16)v;
        else             Ch[(size_t)m * ldc + n] = (_Float16)v;
      }
    }
  }
}

// ---------------------------------------------------------------------------
// Attention: one block per (head, 16 query rows).
// Pass 1: S[16,4096] = q @ k^T / sqrt(64) via WMMA -> 256KB dynamic LDS.
// Softmax stats block-wide; exp() in place, 1/sum folded into pass 2 frags.
// Pass 2: O = P @ V via WMMA, then xt = xn + O (disjoint 16x64 slab).
// K/V panels are shared by 256 blocks per head -> L2-resident streaming.
// ---------------------------------------------------------------------------
__global__ __launch_bounds__(256)
void attn_kernel(const _Float16* __restrict__ q,
                 const _Float16* __restrict__ k,
                 const _Float16* __restrict__ vT,   // [DMODEL][NTOK], head-major
                 const float*    __restrict__ xn,
                 float*          __restrict__ xt)
{
  extern __shared__ float S[];                 // [16][NTOK]  (256 KB)
  __shared__ float red[256];
  __shared__ float rowm[16];
  __shared__ float rowinv[16];
  __shared__ float opart[2 * 16 * DHEAD];      // two K-half partials

  const int h   = blockIdx.y;
  const int qb  = blockIdx.x;
  const int tid = threadIdx.x;
  const int lane = tid & 31, wid = tid >> 5;
  const int row = lane & 15, half = lane >> 4;
  const float scale = 0.125f;                  // 1/sqrt(DHEAD)

  // ---- Pass 1: scores ----
  const _Float16* qbase = q + (size_t)(qb * 16 + row) * DMODEL + h * DHEAD + half * 8;
  for (int kt = wid; kt < NTOK / 16; kt += 8) {
    const _Float16* kbase = k + (size_t)(kt * 16 + row) * DMODEL + h * DHEAD + half * 8;
    v8f acc = {};
    #pragma unroll
    for (int k0 = 0; k0 < DHEAD; k0 += 32)
      acc = wmma_f16(ldfrag(qbase + k0), ldfrag(kbase + k0), acc);
    const int n = kt * 16 + row;
    #pragma unroll
    for (int j = 0; j < 8; ++j)
      S[(size_t)(half * 8 + j) * NTOK + n] = acc[j] * scale;
  }
  __syncthreads();

  // ---- Softmax stats: 16 threads per row ----
  {
    const int r = tid >> 4, c = tid & 15;
    float* Sr = S + (size_t)r * NTOK;
    float lmax = -1e30f;
    for (int i = c; i < NTOK; i += 16) lmax = fmaxf(lmax, Sr[i]);
    red[tid] = lmax; __syncthreads();
    if (c == 0) {
      float m = red[tid];
      for (int i = 1; i < 16; ++i) m = fmaxf(m, red[tid + i]);
      rowm[r] = m;
    }
    __syncthreads();
    const float m = rowm[r];
    float ls = 0.0f;
    for (int i = c; i < NTOK; i += 16) {
      float e = __expf(Sr[i] - m);
      Sr[i] = e;
      ls += e;
    }
    red[tid] = ls; __syncthreads();
    if (c == 0) {
      float s = 0.0f;
      for (int i = 0; i < 16; ++i) s += red[tid + i];
      rowinv[r] = 1.0f / s;
    }
    __syncthreads();
  }

  // ---- Pass 2: O = P @ V ----
  {
    const int t  = wid & 3;       // which 16-wide slice of DHEAD
    const int kh = wid >> 2;      // which 2048-key half
    const _Float16* vbase = vT + (size_t)(h * DHEAD + t * 16 + row) * NTOK + half * 8;
    const float rinv = rowinv[row];
    const float* Sp = S + (size_t)row * NTOK + half * 8;
    v8f acc = {};
    for (int k0 = kh * 2048; k0 < kh * 2048 + 2048; k0 += 32) {
      const float* p = Sp + k0;
      v16h a;
      #pragma unroll
      for (int j = 0; j < 8; ++j) a[j]     = (_Float16)(p[j]      * rinv);
      #pragma unroll
      for (int j = 0; j < 8; ++j) a[8 + j] = (_Float16)(p[16 + j] * rinv);
      acc = wmma_f16(a, ldfrag(vbase + k0), acc);
    }
    const int n = t * 16 + row;
    #pragma unroll
    for (int j = 0; j < 8; ++j)
      opart[kh * (16 * DHEAD) + (half * 8 + j) * DHEAD + n] = acc[j];
  }
  __syncthreads();

  // ---- xt = xn + att for this block's 16x64 slab ----
  for (int e = tid; e < 16 * DHEAD; e += 256) {
    const int m = e >> 6, c = e & 63;
    const float val = opart[e] + opart[16 * DHEAD + e];
    const size_t gi = (size_t)(qb * 16 + m) * DMODEL + h * DHEAD + c;
    xt[gi] = xn[gi] + val;
  }
}

// ---------------------------------------------------------------------------
extern "C" void kernel_launch(void* const* d_in, const int* in_sizes, int n_in,
                              void* d_out, int out_size, void* d_ws, size_t ws_size,
                              hipStream_t stream) {
  (void)in_sizes; (void)n_in; (void)out_size; (void)ws_size;
  const float* x    = (const float*)d_in[0];
  const float* emb  = (const float*)d_in[1];
  const float* Wq   = (const float*)d_in[2];
  const float* Wk   = (const float*)d_in[3];
  const float* Wv   = (const float*)d_in[4];
  const float* g1   = (const float*)d_in[5];
  const float* b1   = (const float*)d_in[6];
  const float* g2   = (const float*)d_in[7];
  const float* b2   = (const float*)d_in[8];
  const float* g3   = (const float*)d_in[9];
  const float* b3   = (const float*)d_in[10];
  const float* W1   = (const float*)d_in[11];
  const float* bf1  = (const float*)d_in[12];
  const float* W2   = (const float*)d_in[13];
  const float* bf2  = (const float*)d_in[14];
  float* out = (float*)d_out;

  char* ws = (char*)d_ws;
  size_t off = 0;
  auto alloc = [&](size_t bytes) -> char* {
    char* p = ws + off;
    off = (off + bytes + 255) & ~(size_t)255;
    return p;
  };
  float*     xn   = (float*)    alloc((size_t)NTOK * DMODEL * 4);
  float*     xt   = (float*)    alloc((size_t)NTOK * DMODEL * 4);
  _Float16*  qkin = (_Float16*) alloc((size_t)NTOK * DQK * 2);
  _Float16*  qh   = (_Float16*) alloc((size_t)NTOK * DMODEL * 2);
  _Float16*  kh   = (_Float16*) alloc((size_t)NTOK * DMODEL * 2);
  _Float16*  vTb  = (_Float16*) alloc((size_t)DMODEL * NTOK * 2);
  _Float16*  hb   = (_Float16*) alloc((size_t)NTOK * DMODEL * 2);
  _Float16*  f1b  = (_Float16*) alloc((size_t)NTOK * DMODEL * 2);
  _Float16*  WqT  = (_Float16*) alloc((size_t)DQK * DMODEL * 2);
  _Float16*  WkT  = (_Float16*) alloc((size_t)DQK * DMODEL * 2);
  _Float16*  WvT  = (_Float16*) alloc((size_t)DMODEL * DMODEL * 2);
  _Float16*  W1T  = (_Float16*) alloc((size_t)DMODEL * DMODEL * 2);
  _Float16*  W2T  = (_Float16*) alloc((size_t)DMODEL * DMODEL * 2);

  // LayerNorms -> fused f16 qk_in [N, 576]; x-part also kept f32 for residual.
  ln_kernel<<<NTOK, DMODEL, 0, stream>>>(x,   DMODEL, g1, b1, qkin,          DQK, xn);
  ln_kernel<<<NTOK, DEMB,   0, stream>>>(emb, DEMB,   g2, b2, qkin + DMODEL, DQK, nullptr);

  // Weight convert + transpose to f16 [out, in] (dense: ldb == K).
  const int cvt_t = 256;
  wcvt_kernel<<<(DQK * DMODEL + cvt_t - 1) / cvt_t, cvt_t, 0, stream>>>(Wq, WqT, DQK, DMODEL);
  wcvt_kernel<<<(DQK * DMODEL + cvt_t - 1) / cvt_t, cvt_t, 0, stream>>>(Wk, WkT, DQK, DMODEL);
  wcvt_kernel<<<(DMODEL * DMODEL + cvt_t - 1) / cvt_t, cvt_t, 0, stream>>>(Wv, WvT, DMODEL, DMODEL);
  wcvt_kernel<<<(DMODEL * DMODEL + cvt_t - 1) / cvt_t, cvt_t, 0, stream>>>(W1, W1T, DMODEL, DMODEL);
  wcvt_kernel<<<(DMODEL * DMODEL + cvt_t - 1) / cvt_t, cvt_t, 0, stream>>>(W2, W2T, DMODEL, DMODEL);

  // Projections. Block tile 128x64, B-panel in LDS via TDM.
  const dim3 gproj(NTOK / 128, DMODEL / 64);
  const size_t ldsK576 = (size_t)64 * DQK    * sizeof(_Float16);  // 72 KB
  const size_t ldsK512 = (size_t)64 * DMODEL * sizeof(_Float16);  // 64 KB
  gemm_kernel<<<gproj, 256, ldsK576, stream>>>(qkin, DQK, WqT, DQK,
                                               nullptr, nullptr, nullptr, qh, DMODEL, 0, 0);
  gemm_kernel<<<gproj, 256, ldsK576, stream>>>(qkin, DQK, WkT, DQK,
                                               nullptr, nullptr, nullptr, kh, DMODEL, 0, 0);
  // v = xn @ Wv, stored transposed as vT[c][n] so PV B-frags are contiguous.
  gemm_kernel<<<gproj, 256, ldsK512, stream>>>(qkin, DQK, WvT, DMODEL,
                                               nullptr, nullptr, nullptr, vTb, NTOK, 1, 0);

  // Attention (+ residual into xt). 256KB dynamic LDS score stripe per block.
  attn_kernel<<<dim3(NTOK / 16, NHEAD), 256, 16 * NTOK * sizeof(float), stream>>>(
      qh, kh, vTb, xn, xt);

  // LN3 -> FFN.
  ln_kernel<<<NTOK, DMODEL, 0, stream>>>(xt, DMODEL, g3, b3, hb, DMODEL, nullptr);
  gemm_kernel<<<gproj, 256, ldsK512, stream>>>(hb, DMODEL, W1T, DMODEL,
                                               bf1, nullptr, nullptr, f1b, DMODEL, 0, 1);
  gemm_kernel<<<gproj, 256, ldsK512, stream>>>(f1b, DMODEL, W2T, DMODEL,
                                               bf2, xt, out, nullptr, DMODEL, 0, 0);
}
